// SwinTransformerBlock3D_v1_63247688401321
// MI455X (gfx1250) — matrix operand order repaired and checked by reading, not compiled, not run on hardware
//
#include <hip/hip_runtime.h>
#include <math.h>

// ---------------------------------------------------------------------------
// Types for CDNA5 WMMA bf16 (16x16x32), wave32.
// ---------------------------------------------------------------------------
typedef __attribute__((ext_vector_type(16))) __bf16 v16bf;
typedef __attribute__((ext_vector_type(8)))  float  v8f;
typedef __attribute__((ext_vector_type(4)))  unsigned int u32x4;

union Frag { v16bf v; u32x4 q[2]; };
union Tile8 { u32x4 q; unsigned short u[8]; };

__device__ __forceinline__ unsigned short f2bf(float f) {
  union { __bf16 b; unsigned short u; } cv;
  cv.b = (__bf16)f;
  return cv.u;
}
__device__ __forceinline__ float bf2f(unsigned short u) {
  union { unsigned short u; __bf16 b; } cv;
  cv.u = u;
  return (float)cv.b;
}

__device__ __forceinline__ v8f vzero8() {
  v8f z;
#pragma unroll
  for (int i = 0; i < 8; ++i) z[i] = 0.f;
  return z;
}

__device__ __forceinline__ v8f wmma_bf16(const Frag& a, const Frag& b, v8f c) {
  // (neg_a, A, neg_b, B, c_mod, C, reuse_a, reuse_b)
  return __builtin_amdgcn_wmma_f32_16x16x32_bf16(false, a.v, false, b.v,
                                                 (short)0, c, false, false);
}

// ---------------------------------------------------------------------------
// Problem constants
// ---------------------------------------------------------------------------
constexpr int MROWS = 50176;   // 128 windows * 392 tokens
constexpr int KT_PAD = 416;    // key dim padded to 13*32
constexpr int NQ_PAD = 400;    // query dim padded to 25*16

// Workspace layout (bytes)
constexpr size_t OFF_WQKV = 0;
constexpr size_t OFF_WPROJ = OFF_WQKV + (size_t)768 * 256 * 2;
constexpr size_t OFF_WFC1  = OFF_WPROJ + (size_t)256 * 256 * 2;
constexpr size_t OFF_WFC2  = OFF_WFC1 + (size_t)1024 * 256 * 2;
constexpr size_t OFF_XW    = OFF_WFC2 + (size_t)256 * 1024 * 2;
constexpr size_t OFF_QT    = OFF_XW + (size_t)MROWS * 256 * 2;          // Q^T [1024][32][400]
constexpr size_t OFF_K     = OFF_QT + (size_t)1024 * 32 * NQ_PAD * 2;   // K   [1024][392][32]
constexpr size_t OFF_VT    = OFF_K + (size_t)MROWS * 256 * 2;           // V^T [1024][32][416]
constexpr size_t OFF_AO    = OFF_VT + (size_t)1024 * 32 * KT_PAD * 2;
constexpr size_t OFF_X2    = OFF_AO + (size_t)MROWS * 256 * 2;
constexpr size_t OFF_H2    = OFF_X2 + (size_t)MROWS * 256 * 4;
constexpr size_t OFF_H1    = OFF_H2 + (size_t)MROWS * 256 * 2;
constexpr size_t OFF_BT    = OFF_H1 + (size_t)MROWS * 1024 * 2;
// bias table: 8 classes x 8 heads x 400 (n) x 416 (m) bf16 = 21.3 MB

// ---------------------------------------------------------------------------
// Weight transpose + f32->bf16: dst[k*Nout + n] = src[n*K + k]
// ---------------------------------------------------------------------------
__global__ __launch_bounds__(256)
void conv_wt(const float* __restrict__ src, unsigned short* __restrict__ dst,
             int Nout, int K) {
  int i = blockIdx.x * 256 + threadIdx.x;
  if (i >= Nout * K) return;
  int k = i / Nout, n = i - k * Nout;
  dst[i] = f2bf(src[(size_t)n * K + k]);
}

// ---------------------------------------------------------------------------
// Zero pad columns [from, from+width) of a strided bf16 matrix (avoids any
// NaN garbage reaching WMMA: 0 x NaN = NaN would poison masked lanes).
// ---------------------------------------------------------------------------
__global__ __launch_bounds__(256)
void zero_pad(unsigned short* __restrict__ p, int stride, int from, int width,
              int rows) {
  int i = blockIdx.x * 256 + threadIdx.x;
  if (i >= rows * width) return;
  int row = i / width, c = i - row * width;
  p[(size_t)row * stride + from + c] = 0;
}

// ---------------------------------------------------------------------------
// Precompute bias+mask table, layout [cls][head][n][m] (m contiguous):
//   relpos_bias(n,m,head) + shift_mask(cls,n,m); m >= 392 -> -1e30.
// cls = (wt==1)*4 | (wh==7)*2 | (ww==7)  (the only 8 distinct window types)
// ---------------------------------------------------------------------------
__global__ __launch_bounds__(256)
void build_bias(const float* __restrict__ rpb, unsigned short* __restrict__ bt) {
  int i = blockIdx.x * 256 + threadIdx.x;  // 8*8*400*416 total
  if (i >= 8 * 8 * NQ_PAD * KT_PAD) return;
  int m = i % KT_PAD; int tmp = i / KT_PAD;
  int n = tmp % NQ_PAD; tmp /= NQ_PAD;
  int head = tmp & 7; int cls = tmp >> 3;

  float val;
  if (m >= 392) {
    val = -1e30f;
  } else if (n >= 392) {
    val = 0.f;
  } else {
    int nt = n / 49, nr = n - nt * 49, nh = nr / 7, nw = nr - nh * 7;
    int mt = m / 49, mr = m - mt * 49, mh = mr / 7, mw = mr - mh * 7;
    int ridx = ((nt - mt + 7) * 169 + (nh - mh + 6) * 13 + (nw - mw + 6)) * 8 + head;
    float bias = rpb[ridx];
    int ft = (cls >> 2) & 1, fh = (cls >> 1) & 1, fw = cls & 1;
    int cn = (ft ? (nt < 4 ? 1 : 2) : 0) * 9 +
             (fh ? (nh < 4 ? 1 : 2) : 0) * 3 +
             (fw ? (nw < 4 ? 1 : 2) : 0);
    int cm = (ft ? (mt < 4 ? 1 : 2) : 0) * 9 +
             (fh ? (mh < 4 ? 1 : 2) : 0) * 3 +
             (fw ? (mw < 4 ? 1 : 2) : 0);
    val = bias + ((cn != cm) ? -100.f : 0.f);
  }
  bt[i] = f2bf(val);
}

// ---------------------------------------------------------------------------
// LN1 + cyclic shift (-4,-3,-3) + window partition -> xw bf16 (MROWS x 256)
// ---------------------------------------------------------------------------
__global__ __launch_bounds__(256)
void ln_shift_part(const float* __restrict__ x, const float* __restrict__ g,
                   const float* __restrict__ b, unsigned short* __restrict__ xw) {
  int lane = threadIdx.x & 31, wid = threadIdx.x >> 5;
  int token = blockIdx.x * 8 + wid;
  int win = token / 392, n = token - win * 392;
  int wt = win >> 6, wh = (win >> 3) & 7, ww = win & 7;
  int lt = n / 49, rem = n - lt * 49, lh = rem / 7, lw = rem - lh * 7;
  int t = (wt * 8 + lt + 4) & 15;
  int h = wh * 7 + lh + 3; if (h >= 56) h -= 56;
  int w = ww * 7 + lw + 3; if (w >= 56) w -= 56;
  const float* src = x + ((size_t)((t * 56 + h) * 56 + w)) * 256;

  float vals[8], sum = 0.f, ss = 0.f;
#pragma unroll
  for (int i = 0; i < 8; ++i) {
    float v = src[lane + 32 * i];
    vals[i] = v; sum += v; ss += v * v;
  }
#pragma unroll
  for (int off = 16; off >= 1; off >>= 1) {
    sum += __shfl_xor(sum, off, 32);
    ss  += __shfl_xor(ss, off, 32);
  }
  float mu = sum * (1.f / 256.f);
  float var = ss * (1.f / 256.f) - mu * mu;
  float rstd = rsqrtf(var + 1e-5f);
  unsigned short* dst = xw + (size_t)token * 256;
#pragma unroll
  for (int i = 0; i < 8; ++i) {
    int c = lane + 32 * i;
    dst[c] = f2bf((vals[i] - mu) * rstd * g[c] + b[c]);
  }
}

// ---------------------------------------------------------------------------
// Plain LN over rows of x2 (spatial order) -> h2 bf16
// ---------------------------------------------------------------------------
__global__ __launch_bounds__(256)
void ln_plain(const float* __restrict__ x2, const float* __restrict__ g,
              const float* __restrict__ b, unsigned short* __restrict__ h2) {
  int lane = threadIdx.x & 31, wid = threadIdx.x >> 5;
  int token = blockIdx.x * 8 + wid;
  const float* src = x2 + (size_t)token * 256;
  float vals[8], sum = 0.f, ss = 0.f;
#pragma unroll
  for (int i = 0; i < 8; ++i) {
    float v = src[lane + 32 * i];
    vals[i] = v; sum += v; ss += v * v;
  }
#pragma unroll
  for (int off = 16; off >= 1; off >>= 1) {
    sum += __shfl_xor(sum, off, 32);
    ss  += __shfl_xor(ss, off, 32);
  }
  float mu = sum * (1.f / 256.f);
  float var = ss * (1.f / 256.f) - mu * mu;
  float rstd = rsqrtf(var + 1e-5f);
  unsigned short* dst = h2 + (size_t)token * 256;
#pragma unroll
  for (int i = 0; i < 8; ++i) {
    int c = lane + 32 * i;
    dst[c] = f2bf((vals[i] - mu) * rstd * g[c] + b[c]);
  }
}

// ---------------------------------------------------------------------------
// Generic bf16 WMMA GEMM: C(M x Nout) = A(M x K) @ Bt(K x Nout) + epilogue.
// Block = 8 waves; each wave a 32x64 tile (2x4 fragments) -> block 128x128.
// ---------------------------------------------------------------------------
enum { EPI_QKV = 0, EPI_PROJ = 1, EPI_FC1 = 2, EPI_FC2 = 3 };

template <int EPI>
__global__ __launch_bounds__(256)
void gemm_k(const unsigned short* __restrict__ A,
            const unsigned short* __restrict__ Bt,
            const float* __restrict__ bias,
            const float* __restrict__ res,    // residual src (x / x2)
            float* __restrict__ outf,         // f32 out (x2 / d_out)
            unsigned short* __restrict__ ob0, // qT / h1
            unsigned short* __restrict__ ob1, // k
            unsigned short* __restrict__ ob2, // vT
            int K, int Nout) {
  int lane = threadIdx.x & 31, wid = threadIdx.x >> 5;
  int half = lane >> 4, l15 = lane & 15;
  int m0 = blockIdx.x * 128 + (wid & 3) * 32;
  int n0 = blockIdx.y * 128 + (wid >> 2) * 64;

  v8f c[2][4];
#pragma unroll
  for (int mt = 0; mt < 2; ++mt)
#pragma unroll
    for (int nt = 0; nt < 4; ++nt) c[mt][nt] = vzero8();

  int rA0 = m0 + l15, rA1 = m0 + 16 + l15;

  for (int kb = 0; kb < K; kb += 32) {
    Frag a[2], b[4];
    const unsigned short* pa0 = A + (size_t)rA0 * K + kb;
    const unsigned short* pa1 = A + (size_t)rA1 * K + kb;
    a[0].q[0] = *(const u32x4*)(pa0 + half * 8);
    a[0].q[1] = *(const u32x4*)(pa0 + 16 + half * 8);
    a[1].q[0] = *(const u32x4*)(pa1 + half * 8);
    a[1].q[1] = *(const u32x4*)(pa1 + 16 + half * 8);
    const unsigned short* pbp = Bt + (size_t)kb * Nout + n0 + l15;
#pragma unroll
    for (int nt = 0; nt < 4; ++nt) {
      Tile8 t0, t1;
#pragma unroll
      for (int s = 0; s < 8; ++s) {
        t0.u[s] = pbp[(size_t)(8 * half + s) * Nout + nt * 16];
        t1.u[s] = pbp[(size_t)(16 + 8 * half + s) * Nout + nt * 16];
      }
      b[nt].q[0] = t0.q;
      b[nt].q[1] = t1.q;
    }
#pragma unroll
    for (int mt = 0; mt < 2; ++mt)
#pragma unroll
      for (int nt = 0; nt < 4; ++nt)
        c[mt][nt] = wmma_bf16(a[mt], b[nt], c[mt][nt]);
  }

#pragma unroll
  for (int mt = 0; mt < 2; ++mt) {
#pragma unroll
    for (int nt = 0; nt < 4; ++nt) {
#pragma unroll
      for (int r = 0; r < 8; ++r) {
        int row = m0 + mt * 16 + r + 8 * half;
        int col = n0 + nt * 16 + l15;
        float val = c[mt][nt][r] + bias[col];
        if constexpr (EPI == EPI_QKV) {
          int which = col >> 8, cmod = col & 255, head = cmod >> 5, d = cmod & 31;
          int win = row / 392, n = row - win * 392;
          size_t ph = (size_t)(win * 8 + head);
          if (which == 0)
            ob0[(ph * 32 + d) * NQ_PAD + n] = f2bf(val * 0.17677669529663687f);
          else if (which == 1)
            ob1[(ph * 392 + n) * 32 + d] = f2bf(val);
          else
            ob2[(ph * 32 + d) * KT_PAD + n] = f2bf(val);
        } else if constexpr (EPI == EPI_PROJ) {
          int win = row / 392, n = row - win * 392;
          int wt = win >> 6, wh = (win >> 3) & 7, ww = win & 7;
          int lt = n / 49, rem = n - lt * 49, lh = rem / 7, lw = rem - lh * 7;
          int t = (wt * 8 + lt + 4) & 15;
          int h = wh * 7 + lh + 3; if (h >= 56) h -= 56;
          int w = ww * 7 + lw + 3; if (w >= 56) w -= 56;
          size_t idx = ((size_t)((t * 56 + h) * 56 + w)) * 256 + col;
          outf[idx] = res[idx] + val;
        } else if constexpr (EPI == EPI_FC1) {
          float gv = 0.5f * val * (1.0f + erff(val * 0.70710678118654752f));
          ob0[(size_t)row * Nout + col] = f2bf(gv);
        } else {  // EPI_FC2
          size_t idx = (size_t)row * 256 + col;
          outf[idx] = res[idx] + val;
        }
      }
    }
  }
}

// ---------------------------------------------------------------------------
// Flash attention, TRANSPOSED: S^T = K @ Q^T, O^T = V^T @ P.
// In the C-fragment layout the column is the query (lane&15), so the softmax
// running state (m, l, alpha) is per-lane: only 1 shfl_xor(16) per iteration.
// One wave per 16-query tile; 13 iterations of 32 keys.
// ---------------------------------------------------------------------------
__global__ __launch_bounds__(256)
void attn_k(const unsigned short* __restrict__ qT,
            const unsigned short* __restrict__ k,
            const unsigned short* __restrict__ vT,
            const unsigned short* __restrict__ bt,
            unsigned short* __restrict__ ao) {
  __shared__ unsigned short plds[8 * 512];  // 32(key) x 16(query) bf16 per wave
  int lane = threadIdx.x & 31, wid = threadIdx.x >> 5;
  int half = lane >> 4, l15 = lane & 15;
  int pair = blockIdx.x >> 2;               // win*8 + head
  int qt = (blockIdx.x & 3) * 8 + wid;
  if (qt >= 25) return;
  int win = pair >> 3, head = pair & 7;
  int wt = win >> 6, wh = (win >> 3) & 7, ww = win & 7;
  int cls = ((wt == 1) ? 4 : 0) | ((wh == 7) ? 2 : 0) | ((ww == 7) ? 1 : 0);
  int nq = qt * 16 + l15;                   // this lane's query token

  // bias row for (cls, head, query n); keys m contiguous
  const unsigned short* btb =
      bt + ((size_t)(cls * 8 + head) * NQ_PAD + nq) * KT_PAD;

  // Q^T B-fragment: lane = query column, elements = head-dim d (table order)
  Frag qf;
  {
    const unsigned short* pq = qT + (size_t)pair * 32 * NQ_PAD + qt * 16 + l15;
    Tile8 t0, t1;
#pragma unroll
    for (int s = 0; s < 8; ++s) {
      t0.u[s] = pq[(size_t)(8 * half + s) * NQ_PAD];
      t1.u[s] = pq[(size_t)(16 + 8 * half + s) * NQ_PAD];
    }
    qf.q[0] = t0.q;
    qf.q[1] = t1.q;
  }
  // V^T A-fragment row bases (d = l15 and 16+l15)
  const unsigned short* pv0 = vT + ((size_t)pair * 32 + l15) * KT_PAD;
  const unsigned short* pv1 = vT + ((size_t)pair * 32 + 16 + l15) * KT_PAD;

  v8f o0 = vzero8(), o1 = vzero8();
  float mrow = -1e30f, lpart = 0.f;

  unsigned short* myp = plds + wid * 512;

  for (int kp = 0; kp < 13; ++kp) {
    int kb = kp * 32;
    // K A-fragments: rows = keys (clamped; masked via bias -1e30)
    Frag ak0, ak1;
    {
      int k0r = kb + l15;      if (k0r > 391) k0r = 391;
      int k1r = kb + 16 + l15; if (k1r > 391) k1r = 391;
      const unsigned short* pk0 = k + ((size_t)pair * 392 + k0r) * 32;
      const unsigned short* pk1 = k + ((size_t)pair * 392 + k1r) * 32;
      ak0.q[0] = *(const u32x4*)(pk0 + half * 8);
      ak0.q[1] = *(const u32x4*)(pk0 + 16 + half * 8);
      ak1.q[0] = *(const u32x4*)(pk1 + half * 8);
      ak1.q[1] = *(const u32x4*)(pk1 + 16 + half * 8);
    }
    // bias tiles: this lane's keys are kb+8*half+r (s0) and +16 (s1)
    Tile8 b0, b1;
    b0.q = *(const u32x4*)(btb + kb + half * 8);
    b1.q = *(const u32x4*)(btb + kb + 16 + half * 8);

    v8f s0 = wmma_bf16(ak0, qf, vzero8());
    v8f s1 = wmma_bf16(ak1, qf, vzero8());

    float xs[16];
#pragma unroll
    for (int r = 0; r < 8; ++r) {
      xs[r]     = s0[r] + bf2f(b0.u[r]);
      xs[8 + r] = s1[r] + bf2f(b1.u[r]);
    }
    // in-lane max over 16 keys, then combine the two key-halves
    float lm = xs[0];
#pragma unroll
    for (int i = 1; i < 16; ++i) lm = fmaxf(lm, xs[i]);
    lm = fmaxf(lm, __shfl_xor(lm, 16, 32));
    float mnew = fmaxf(mrow, lm);
    float alpha = __expf(mrow - mnew);
    mrow = mnew;

    float ps[16], ls = 0.f;
#pragma unroll
    for (int i = 0; i < 16; ++i) {
      ps[i] = __expf(xs[i] - mnew);
      ls += ps[i];
    }
    lpart = lpart * alpha + ls;   // cross-half combined once at the end
    o0 = o0 * alpha;
    o1 = o1 * alpha;

    // store P to LDS as [query][key]
#pragma unroll
    for (int r = 0; r < 8; ++r) {
      myp[l15 * 32 + r + 8 * half]      = f2bf(ps[r]);
      myp[l15 * 32 + 16 + r + 8 * half] = f2bf(ps[8 + r]);
    }
    asm volatile("s_wait_dscnt 0" ::: "memory");  // per-wave LDS RAW fence

    // P B-fragment: lane = query column, elements = keys (table order)
    Frag pB;
    {
      const unsigned short* pl = myp + l15 * 32;
      pB.q[0] = *(const u32x4*)(pl + 8 * half);
      pB.q[1] = *(const u32x4*)(pl + 16 + 8 * half);
    }
    // V^T A-fragments over this key chunk
    Frag av0, av1;
    av0.q[0] = *(const u32x4*)(pv0 + kb + half * 8);
    av0.q[1] = *(const u32x4*)(pv0 + kb + 16 + half * 8);
    av1.q[0] = *(const u32x4*)(pv1 + kb + half * 8);
    av1.q[1] = *(const u32x4*)(pv1 + kb + 16 + half * 8);

    o0 = wmma_bf16(av0, pB, o0);
    o1 = wmma_bf16(av1, pB, o1);
  }

  float l = lpart + __shfl_xor(lpart, 16, 32);
  if (nq < 392) {
    float inv = 1.0f / l;
    size_t base = ((size_t)win * 392 + nq) * 256 + head * 32;
    Tile8 t0, t1;
#pragma unroll
    for (int r = 0; r < 8; ++r) {
      t0.u[r] = f2bf(o0[r] * inv);   // d = 8*half + r
      t1.u[r] = f2bf(o1[r] * inv);   // d = 16 + 8*half + r
    }
    *(u32x4*)(ao + base + 8 * half) = t0.q;
    *(u32x4*)(ao + base + 16 + 8 * half) = t1.q;
  }
}

// ---------------------------------------------------------------------------
// Launch
// ---------------------------------------------------------------------------
extern "C" void kernel_launch(void* const* d_in, const int* in_sizes, int n_in,
                              void* d_out, int out_size, void* d_ws, size_t ws_size,
                              hipStream_t stream) {
  (void)in_sizes; (void)n_in; (void)out_size; (void)ws_size;
  const float* x    = (const float*)d_in[0];
  const float* n1g  = (const float*)d_in[1];
  const float* n1b  = (const float*)d_in[2];
  const float* qkvw = (const float*)d_in[3];
  const float* qkvb = (const float*)d_in[4];
  const float* rpb  = (const float*)d_in[5];
  const float* pw   = (const float*)d_in[6];
  const float* pb   = (const float*)d_in[7];
  const float* n2g  = (const float*)d_in[8];
  const float* n2b  = (const float*)d_in[9];
  const float* f1w  = (const float*)d_in[10];
  const float* f1b  = (const float*)d_in[11];
  const float* f2w  = (const float*)d_in[12];
  const float* f2b  = (const float*)d_in[13];

  char* ws = (char*)d_ws;
  unsigned short* wqkv  = (unsigned short*)(ws + OFF_WQKV);
  unsigned short* wproj = (unsigned short*)(ws + OFF_WPROJ);
  unsigned short* wfc1  = (unsigned short*)(ws + OFF_WFC1);
  unsigned short* wfc2  = (unsigned short*)(ws + OFF_WFC2);
  unsigned short* xw    = (unsigned short*)(ws + OFF_XW);
  unsigned short* qtb   = (unsigned short*)(ws + OFF_QT);
  unsigned short* kb    = (unsigned short*)(ws + OFF_K);
  unsigned short* vtb   = (unsigned short*)(ws + OFF_VT);
  unsigned short* ao    = (unsigned short*)(ws + OFF_AO);
  float*          x2    = (float*)(ws + OFF_X2);
  unsigned short* h2    = (unsigned short*)(ws + OFF_H2);
  unsigned short* h1    = (unsigned short*)(ws + OFF_H1);
  unsigned short* bt    = (unsigned short*)(ws + OFF_BT);
  float* out = (float*)d_out;

  // One-time prep: weight transpose/convert, pad zeroing, bias+mask table
  conv_wt<<<(768 * 256 + 255) / 256, 256, 0, stream>>>(qkvw, wqkv, 768, 256);
  conv_wt<<<(256 * 256 + 255) / 256, 256, 0, stream>>>(pw, wproj, 256, 256);
  conv_wt<<<(1024 * 256 + 255) / 256, 256, 0, stream>>>(f1w, wfc1, 1024, 256);
  conv_wt<<<(256 * 1024 + 255) / 256, 256, 0, stream>>>(f2w, wfc2, 256, 1024);
  zero_pad<<<(1024 * 32 * 8 + 255) / 256, 256, 0, stream>>>(
      qtb, NQ_PAD, 392, 8, 1024 * 32);
  zero_pad<<<(1024 * 32 * 24 + 255) / 256, 256, 0, stream>>>(
      vtb, KT_PAD, 392, 24, 1024 * 32);
  build_bias<<<(8 * 8 * NQ_PAD * KT_PAD + 255) / 256, 256, 0, stream>>>(rpb, bt);

  // LN1 + shift + window partition
  ln_shift_part<<<MROWS / 8, 256, 0, stream>>>(x, n1g, n1b, xw);

  // QKV GEMM (50176x256 @ 256x768) -> Q^T (scaled), K, V^T
  gemm_k<EPI_QKV><<<dim3(392, 6), 256, 0, stream>>>(
      xw, wqkv, qkvb, nullptr, nullptr, qtb, kb, vtb, 256, 768);

  // Flash attention: 1024 (win,head) pairs * 4 blocks, 8 waves each
  attn_k<<<4096, 256, 0, stream>>>(qtb, kb, vtb, bt, ao);

  // Proj GEMM + window-reverse + unshift + residual -> x2 (f32, spatial)
  gemm_k<EPI_PROJ><<<dim3(392, 2), 256, 0, stream>>>(
      ao, wproj, pb, x, x2, nullptr, nullptr, nullptr, 256, 256);

  // LN2
  ln_plain<<<MROWS / 8, 256, 0, stream>>>(x2, n2g, n2b, h2);

  // fc1 + GELU
  gemm_k<EPI_FC1><<<dim3(392, 8), 256, 0, stream>>>(
      h2, wfc1, f1b, nullptr, nullptr, h1, nullptr, nullptr, 256, 1024);

  // fc2 + residual -> d_out
  gemm_k<EPI_FC2><<<dim3(392, 2), 256, 0, stream>>>(
      h1, wfc2, f2b, x2, out, nullptr, nullptr, nullptr, 1024, 256);
}
